// Encoder_23467701305748
// MI455X (gfx1250) — matrix operand order repaired and checked by reading, not compiled, not run on hardware
//
#include <hip/hip_runtime.h>

// ---------------------------------------------------------------------------
// Problem constants (match reference)
// ---------------------------------------------------------------------------
#define BB 32
#define SS 512
#define DD 512
#define HH 16
#define DH 32
#define FF 2048
#define LL 3
#define ROWS (BB * SS)          // 16384
#define NTOT (ROWS * DD)        // 8388608
#define EPSV 1e-3f

// ---------------------------------------------------------------------------
// Types for WMMA
// ---------------------------------------------------------------------------
typedef __attribute__((ext_vector_type(16))) __bf16      bf16x16;
typedef __attribute__((ext_vector_type(8)))  float       f32x8;
typedef __attribute__((ext_vector_type(4)))  unsigned int u32x4;

union Frag { bf16x16 v; u32x4 q[2]; };

__device__ __forceinline__ unsigned short f2bf(float f) {
    unsigned int u = __float_as_uint(f);
    u += 0x7fffu + ((u >> 16) & 1u);     // round-to-nearest-even
    return (unsigned short)(u >> 16);
}

__device__ __forceinline__ u32x4 ld16(const unsigned short* p) {
    return *(const u32x4*)p;
}

__device__ __forceinline__ f32x8 wmma_bf16(const Frag& a, const Frag& b, f32x8 c) {
    return __builtin_amdgcn_wmma_f32_16x16x32_bf16(false, a.v, false, b.v,
                                                   (short)0, c, false, false);
}

// ---------------------------------------------------------------------------
// Generic WMMA GEMM: out = act(A[M,K] @ Bt[N,K]^T + bias[N])
//   A  : bf16 row-major [M][K]
//   Bt : bf16 row-major [N][K]  (i.e. W transposed)
// mode 0: fp32 row-major out, no relu          (FFN2)
// mode 1: bf16 row-major out, relu             (FFN1)
// mode 2: bf16 out at [b][h][s][dh], relu      (Q,K)
// mode 3: bf16 out at [b][h][dh][s], relu      (V, pre-transposed for PV gemm)
// Block: 256 thr = 8 waves (4Mx2N), wave tile 32x32, block tile 128x64.
// ---------------------------------------------------------------------------
__global__ __launch_bounds__(256) void gemm_wmma(
    const unsigned short* __restrict__ A, const unsigned short* __restrict__ Bt,
    const float* __restrict__ bias, void* __restrict__ out,
    int M, int N, int K, int mode)
{
    const int lane = threadIdx.x & 31;
    const int wave = threadIdx.x >> 5;
    const int lo   = lane & 15;
    const int half = lane >> 4;
    const int m0 = blockIdx.x * 128 + (wave >> 1) * 32;
    const int n0 = blockIdx.y * 64  + (wave & 1) * 32;

    const unsigned short* a0p = A  + (size_t)(m0 + lo)      * K;
    const unsigned short* a1p = A  + (size_t)(m0 + 16 + lo) * K;
    const unsigned short* b0p = Bt + (size_t)(n0 + lo)      * K;
    const unsigned short* b1p = Bt + (size_t)(n0 + 16 + lo) * K;

    const f32x8 z8 = {0.f,0.f,0.f,0.f,0.f,0.f,0.f,0.f};
    f32x8 c00 = z8, c01 = z8, c10 = z8, c11 = z8;

    for (int k = 0; k < K; k += 32) {
        Frag a0, a1, b0, b1;
        // A frag (16-bit 16x32): lane holds row m, 16 contiguous K at 16*half
        a0.q[0] = ld16(a0p + k + 16 * half);
        a0.q[1] = ld16(a0p + k + 16 * half + 8);
        a1.q[0] = ld16(a1p + k + 16 * half);
        a1.q[1] = ld16(a1p + k + 16 * half + 8);
        // B frag (16-bit 32x16): lane holds column n, 16 contiguous K at 16*half
        b0.q[0] = ld16(b0p + k + 16 * half);
        b0.q[1] = ld16(b0p + k + 16 * half + 8);
        b1.q[0] = ld16(b1p + k + 16 * half);
        b1.q[1] = ld16(b1p + k + 16 * half + 8);

        c00 = wmma_bf16(a0, b0, c00);
        c01 = wmma_bf16(a0, b1, c01);
        c10 = wmma_bf16(a1, b0, c10);
        c11 = wmma_bf16(a1, b1, c11);
    }

    f32x8 acc[2][2] = {{c00, c01}, {c10, c11}};
    #pragma unroll
    for (int i = 0; i < 2; ++i) {
        #pragma unroll
        for (int j = 0; j < 2; ++j) {
            const int nc = n0 + j * 16 + lo;
            const float bv = bias[nc];
            #pragma unroll
            for (int r = 0; r < 8; ++r) {
                float vf = acc[i][j][r] + bv;
                if (mode != 0) vf = fmaxf(vf, 0.f);
                const int row = m0 + i * 16 + r + 8 * half;   // C row = r + 8*half
                if (mode == 0) {
                    ((float*)out)[(size_t)row * N + nc] = vf;
                } else if (mode == 1) {
                    ((unsigned short*)out)[(size_t)row * N + nc] = f2bf(vf);
                } else {
                    const int b = row >> 9, s = row & 511;
                    const int h = nc >> 5,  dh = nc & 31;
                    size_t idx;
                    if (mode == 2) idx = (((size_t)(b * HH + h)) * SS + s) * DH + dh;
                    else           idx = (((size_t)(b * HH + h)) * DH + dh) * SS + s;
                    ((unsigned short*)out)[idx] = f2bf(vf);
                }
            }
        }
    }
}

// ---------------------------------------------------------------------------
// Fused attention: per wave = one 16-row strip of one (b,h).
//  pass1: QK^T (1 WMMA per 16-col tile, K=DH=32), track row max
//  pass2: recompute, exp, bf16 P -> LDS, track row sum
//  pass3: O = P @ V^T from LDS, scale by 1/rowsum in C-frag
// Block = 64 threads (2 waves), LDS = 2 * 16*512 bf16 = 32 KB.
// ---------------------------------------------------------------------------
__global__ __launch_bounds__(64) void attn_kernel(
    const unsigned short* __restrict__ qb, const unsigned short* __restrict__ kb,
    const unsigned short* __restrict__ vtb, float* __restrict__ o)
{
    __shared__ unsigned short pl[2 * 16 * SS];

    const int lane = threadIdx.x & 31;
    const int wave = threadIdx.x >> 5;
    const int lo   = lane & 15;
    const int half = lane >> 4;

    const int strip = blockIdx.x * 2 + wave;       // 16384 strips total
    const int bh = strip >> 5;                     // (b*H + h)
    const int m0 = (strip & 31) << 4;              // row tile within S
    const int b  = bh >> 4;
    const int h  = bh & 15;

    const unsigned short* qp = qb  + (size_t)bh * SS * DH;
    const unsigned short* kp = kb  + (size_t)bh * SS * DH;
    const unsigned short* vp = vtb + (size_t)bh * DH * SS;
    unsigned short* myp = &pl[wave * 16 * SS];

    // Q A-frag: rows m0..m0+15, K = dh 0..31 (one frag covers DH)
    Frag qa;
    {
        const unsigned short* p = qp + (size_t)(m0 + lo) * DH;
        qa.q[0] = ld16(p + 16 * half);
        qa.q[1] = ld16(p + 16 * half + 8);
    }

    const f32x8 z8 = {0.f,0.f,0.f,0.f,0.f,0.f,0.f,0.f};
    const float scale = 0.17677669529663687f;      // 1/sqrt(32)

    float rmax[8];
    #pragma unroll
    for (int r = 0; r < 8; ++r) rmax[r] = -3.0e38f;

    // pass 1: row max of raw scores
    for (int t0 = 0; t0 < SS; t0 += 16) {
        Frag kf;
        const unsigned short* p = kp + (size_t)(t0 + lo) * DH + 16 * half;
        kf.q[0] = ld16(p);
        kf.q[1] = ld16(p + 8);
        f32x8 c = wmma_bf16(qa, kf, z8);
        #pragma unroll
        for (int r = 0; r < 8; ++r) rmax[r] = fmaxf(rmax[r], c[r]);
    }
    #pragma unroll
    for (int m = 1; m < 16; m <<= 1) {
        #pragma unroll
        for (int r = 0; r < 8; ++r)
            rmax[r] = fmaxf(rmax[r], __shfl_xor(rmax[r], m, 32));
    }

    // pass 2: exp, store unnormalized P (bf16) to LDS, row sums
    float rsum[8];
    #pragma unroll
    for (int r = 0; r < 8; ++r) rsum[r] = 0.f;

    for (int t0 = 0; t0 < SS; t0 += 16) {
        Frag kf;
        const unsigned short* p = kp + (size_t)(t0 + lo) * DH + 16 * half;
        kf.q[0] = ld16(p);
        kf.q[1] = ld16(p + 8);
        f32x8 c = wmma_bf16(qa, kf, z8);
        #pragma unroll
        for (int r = 0; r < 8; ++r) {
            const float e = __expf((c[r] - rmax[r]) * scale);
            rsum[r] += e;
            myp[(r + 8 * half) * SS + t0 + lo] = f2bf(e);
        }
    }
    #pragma unroll
    for (int m = 1; m < 16; m <<= 1) {
        #pragma unroll
        for (int r = 0; r < 8; ++r) rsum[r] += __shfl_xor(rsum[r], m, 32);
    }
    float rinv[8];
    #pragma unroll
    for (int r = 0; r < 8; ++r) rinv[r] = 1.f / rsum[r];

    // pass 3: O = P @ V^T  (N = DH = 32 -> two 16-col tiles, K = S = 512)
    #pragma unroll
    for (int nt = 0; nt < 2; ++nt) {
        f32x8 acc = z8;
        for (int t0 = 0; t0 < SS; t0 += 32) {
            Frag pa, vf;
            const unsigned short* lp = myp + lo * SS + t0;
            pa.q[0] = ld16(lp + 16 * half);
            pa.q[1] = ld16(lp + 16 * half + 8);
            const unsigned short* vpp = vp + (size_t)(nt * 16 + lo) * SS + t0 + 16 * half;
            vf.q[0] = ld16(vpp);
            vf.q[1] = ld16(vpp + 8);
            acc = wmma_bf16(pa, vf, acc);
        }
        #pragma unroll
        for (int r = 0; r < 8; ++r) {
            const int row = m0 + r + 8 * half;
            o[(((size_t)(b * SS + row)) * HH + h) * DH + nt * 16 + lo] = acc[r] * rinv[r];
        }
    }
}

// ---------------------------------------------------------------------------
// Helpers: weight transpose+convert, input prep, batchnorm
// ---------------------------------------------------------------------------
__global__ void tconv_kernel(const float* __restrict__ src,
                             unsigned short* __restrict__ dst, int R, int C)
{
    const int n = R * C;
    for (int i = blockIdx.x * blockDim.x + threadIdx.x; i < n;
         i += gridDim.x * blockDim.x) {
        const int r = i / C, c = i - r * C;
        dst[(size_t)c * R + r] = f2bf(src[i]);       // dst[n][k] = src[k][n]
    }
}

__global__ void xprep_kernel(const float* __restrict__ src, float* __restrict__ xf,
                             unsigned short* __restrict__ xb, int n)
{
    for (int i = blockIdx.x * blockDim.x + threadIdx.x; i < n;
         i += gridDim.x * blockDim.x) {
        const float v = src[i];
        xf[i] = v;
        xb[i] = f2bf(v);
    }
}

__global__ void zero_kernel(float* __restrict__ p)
{
    p[threadIdx.x] = 0.f;
}

__global__ __launch_bounds__(256) void bn_stats_kernel(
    const float* __restrict__ a, const float* __restrict__ res,
    float* __restrict__ sums, int n)
{
    __shared__ float ls[DD];
    __shared__ float lq[DD];
    for (int i = threadIdx.x; i < DD; i += 256) { ls[i] = 0.f; lq[i] = 0.f; }
    __syncthreads();
    for (int i = blockIdx.x * 256 + threadIdx.x; i < n; i += gridDim.x * 256) {
        const float v = a[i] + res[i];
        const int c = i & (DD - 1);
        atomicAdd(&ls[c], v);
        atomicAdd(&lq[c], v * v);
    }
    __syncthreads();
    for (int i = threadIdx.x; i < DD; i += 256) {
        atomicAdd(&sums[i], ls[i]);
        atomicAdd(&sums[DD + i], lq[i]);
    }
}

__global__ __launch_bounds__(256) void bn_apply_kernel(
    const float* __restrict__ a, const float* __restrict__ res,
    const float* __restrict__ sums, const float* __restrict__ g,
    const float* __restrict__ be, float* __restrict__ xout,
    unsigned short* __restrict__ xbout, int n)
{
    const float inv = 1.f / (float)ROWS;
    for (int i = blockIdx.x * 256 + threadIdx.x; i < n; i += gridDim.x * 256) {
        const int c = i & (DD - 1);
        const float mean = sums[c] * inv;
        const float var  = sums[DD + c] * inv - mean * mean;
        const float rstd = rsqrtf(var + EPSV);
        const float v = a[i] + res[i];
        const float y = (v - mean) * rstd * g[c] + be[c];
        xout[i]  = y;
        xbout[i] = f2bf(y);
    }
}

// ---------------------------------------------------------------------------
// Host: orchestrate one full encoder forward
// ---------------------------------------------------------------------------
extern "C" void kernel_launch(void* const* d_in, const int* in_sizes, int n_in,
                              void* d_out, int out_size, void* d_ws, size_t ws_size,
                              hipStream_t stream)
{
    (void)in_sizes; (void)n_in; (void)out_size; (void)ws_size;

    const float* input = (const float*)d_in[0];
    const float* Wq = (const float*)d_in[1];  const float* bq = (const float*)d_in[2];
    const float* Wk = (const float*)d_in[3];  const float* bk = (const float*)d_in[4];
    const float* Wv = (const float*)d_in[5];  const float* bv = (const float*)d_in[6];
    const float* g1 = (const float*)d_in[7];  const float* be1 = (const float*)d_in[8];
    const float* W1 = (const float*)d_in[9];  const float* b1 = (const float*)d_in[10];
    const float* W2 = (const float*)d_in[11]; const float* b2 = (const float*)d_in[12];
    const float* g2 = (const float*)d_in[13]; const float* be2 = (const float*)d_in[14];
    float* outp = (float*)d_out;

    // --- workspace layout ---
    char* ws = (char*)d_ws;
    size_t off = 0;
    float*          xf  = (float*)(ws + off);          off += (size_t)NTOT * 4;   // 33.5MB
    unsigned short* xb  = (unsigned short*)(ws + off); off += (size_t)NTOT * 2;   // 16.8MB
    unsigned short* qb  = (unsigned short*)(ws + off); off += (size_t)NTOT * 2;
    unsigned short* kbuf= (unsigned short*)(ws + off); off += (size_t)NTOT * 2;
    unsigned short* vtb = (unsigned short*)(ws + off); off += (size_t)NTOT * 2;
    float*          obuf= (float*)(ws + off);          off += (size_t)NTOT * 4;   // 33.5MB
    unsigned short* hb  = (unsigned short*)(ws + off); off += (size_t)ROWS * FF * 2; // 67MB
    unsigned short* wqT = (unsigned short*)(ws + off); off += (size_t)LL * DD * DD * 2;
    unsigned short* wkT = (unsigned short*)(ws + off); off += (size_t)LL * DD * DD * 2;
    unsigned short* wvT = (unsigned short*)(ws + off); off += (size_t)LL * DD * DD * 2;
    unsigned short* w1T = (unsigned short*)(ws + off); off += (size_t)LL * DD * FF * 2;
    unsigned short* w2T = (unsigned short*)(ws + off); off += (size_t)LL * FF * DD * 2;
    float*          sums= (float*)(ws + off);          off += 2 * DD * 4;

    // --- prep: input copy/convert, weight transpose+convert ---
    xprep_kernel<<<2048, 256, 0, stream>>>(input, xf, xb, NTOT);
    for (int i = 0; i < LL; ++i) {
        tconv_kernel<<<1024, 256, 0, stream>>>(Wq + (size_t)i * DD * DD, wqT + (size_t)i * DD * DD, DD, DD);
        tconv_kernel<<<1024, 256, 0, stream>>>(Wk + (size_t)i * DD * DD, wkT + (size_t)i * DD * DD, DD, DD);
        tconv_kernel<<<1024, 256, 0, stream>>>(Wv + (size_t)i * DD * DD, wvT + (size_t)i * DD * DD, DD, DD);
        tconv_kernel<<<4096, 256, 0, stream>>>(W1 + (size_t)i * DD * FF, w1T + (size_t)i * DD * FF, DD, FF);
        tconv_kernel<<<4096, 256, 0, stream>>>(W2 + (size_t)i * FF * DD, w2T + (size_t)i * FF * DD, FF, DD);
    }

    const dim3 gqkv(ROWS / 128, DD / 64);   // 128 x 8
    const dim3 gff1(ROWS / 128, FF / 64);   // 128 x 32
    const dim3 gff2(ROWS / 128, DD / 64);   // 128 x 8

    for (int i = 0; i < LL; ++i) {
        // --- attention block ---
        gemm_wmma<<<gqkv, 256, 0, stream>>>(xb, wqT + (size_t)i * DD * DD, bq + (size_t)i * DD,
                                            qb,   ROWS, DD, DD, 2);
        gemm_wmma<<<gqkv, 256, 0, stream>>>(xb, wkT + (size_t)i * DD * DD, bk + (size_t)i * DD,
                                            kbuf, ROWS, DD, DD, 2);
        gemm_wmma<<<gqkv, 256, 0, stream>>>(xb, wvT + (size_t)i * DD * DD, bv + (size_t)i * DD,
                                            vtb,  ROWS, DD, DD, 3);
        attn_kernel<<<BB * HH * (SS / 16) / 2, 64, 0, stream>>>(qb, kbuf, vtb, obuf);

        zero_kernel<<<1, 2 * DD, 0, stream>>>(sums);
        bn_stats_kernel<<<512, 256, 0, stream>>>(obuf, xf, sums, NTOT);
        bn_apply_kernel<<<2048, 256, 0, stream>>>(obuf, xf, sums,
                                                  g1 + (size_t)i * DD, be1 + (size_t)i * DD,
                                                  xf, xb, NTOT);

        // --- feed-forward block ---
        gemm_wmma<<<gff1, 256, 0, stream>>>(xb, w1T + (size_t)i * DD * FF, b1 + (size_t)i * FF,
                                            hb,   ROWS, FF, DD, 1);
        gemm_wmma<<<gff2, 256, 0, stream>>>(hb, w2T + (size_t)i * FF * DD, b2 + (size_t)i * DD,
                                            obuf, ROWS, DD, FF, 0);

        zero_kernel<<<1, 2 * DD, 0, stream>>>(sums);
        bn_stats_kernel<<<512, 256, 0, stream>>>(obuf, xf, sums, NTOT);
        float* xdst = (i == LL - 1) ? outp : xf;
        bn_apply_kernel<<<2048, 256, 0, stream>>>(obuf, xf, sums,
                                                  g2 + (size_t)i * DD, be2 + (size_t)i * DD,
                                                  xdst, xb, NTOT);
    }
}
